// CustomModel_52484500357175
// MI455X (gfx1250) — matrix operand order repaired and checked by reading, not compiled, not run on hardware
//
#include <hip/hip_runtime.h>
#include <hip/hip_bf16.h>
#include <stdint.h>

#define HH      512
#define OUTC    300
#define OUTPAD  384     // ceil(300/128)*128: output-head weight padded so B loads never leave the allocation
#define EPSB    1e-5f

typedef __attribute__((ext_vector_type(16))) __bf16 v16bf;
typedef __attribute__((ext_vector_type(8)))  float  v8f;
typedef unsigned short ushort_t;

// ---------- helpers ----------
__device__ __forceinline__ ushort_t f32_bf16(float f) {
    union { float f; uint32_t u; } x; x.f = f;
    uint32_t u = x.u;
    u += 0x7FFFu + ((u >> 16) & 1u);          // round-to-nearest-even
    return (ushort_t)(u >> 16);
}

union FragU { v16bf v; int4 q[2]; };

// ---------- weight convert + transpose: in[K=512, Ncols] f32 -> out[gridDim.x, 512] bf16 ----------
__global__ __launch_bounds__(256)
void gcn_convT_kernel(const float* __restrict__ in, ushort_t* __restrict__ out, int Ncols) {
    int n = blockIdx.x;                        // output row = original column (0..NPad-1)
    for (int k = threadIdx.x; k < HH; k += 256) {
        float v = (n < Ncols) ? in[(size_t)k * Ncols + n] : 0.f;
        out[(size_t)n * HH + k] = f32_bf16(v);
    }
}

// ---------- layer-1 rank-1 scatter: s[dst] += x[src] ----------
__global__ __launch_bounds__(256)
void gcn_scatter_scalar_kernel(const float* __restrict__ x, const int* __restrict__ src,
                               const int* __restrict__ dst, float* __restrict__ s, int E) {
    int e = blockIdx.x * 256 + threadIdx.x;
    if (e < E) atomicAdd(&s[dst[e]], x[src[e]]);
}

// ---------- scalar sum/sumsq reduction of s ----------
__global__ __launch_bounds__(256)
void gcn_reduce_scalar_kernel(const float* __restrict__ s, float* __restrict__ out2, int N) {
    float a = 0.f, b = 0.f;
    for (int i = blockIdx.x * 256 + threadIdx.x; i < N; i += gridDim.x * 256) {
        float v = s[i]; a += v; b += v * v;
    }
    __shared__ float sa[256], sb[256];
    sa[threadIdx.x] = a; sb[threadIdx.x] = b; __syncthreads();
    for (int o = 128; o > 0; o >>= 1) {
        if (threadIdx.x < o) { sa[threadIdx.x] += sa[threadIdx.x + o]; sb[threadIdx.x] += sb[threadIdx.x + o]; }
        __syncthreads();
    }
    if (threadIdx.x == 0) { atomicAdd(&out2[0], sa[0]); atomicAdd(&out2[1], sb[0]); }
}

// ---------- alter branch part a: out[N,512]bf16 = alt[N,6] @ W6[6,512] (bias cancels in BN) ----------
__global__ __launch_bounds__(256)
void gcn_alter_mm_kernel(const float* __restrict__ alt, const float* __restrict__ W6,
                         ushort_t* __restrict__ out) {
    int r = blockIdx.x;
    __shared__ float arow[8];
    if (threadIdx.x < 6) arow[threadIdx.x] = alt[(size_t)r * 6 + threadIdx.x];
    __syncthreads();
    for (int c = threadIdx.x; c < HH; c += 256) {
        float acc = 0.f;
        #pragma unroll
        for (int k = 0; k < 6; ++k) acc += arow[k] * W6[k * HH + c];
        out[(size_t)r * HH + c] = f32_bf16(acc);
    }
}

// ---------- WMMA GEMM: C[M,ldc] = A[M,512]bf16 @ Bt[Npad,512]bf16^T (+bias), f32 accumulate ----------
// Register-blocked: each wave owns a 2x4 grid of 16x16 tiles (32 rows x 64 cols); per K-step of 32,
// 2 A-fragments + 4 B-fragments feed 8 v_wmma_f32_16x16x32_bf16 -> ~21 FLOP per L2 byte.
// Block = 8 waves in 4(M)x2(N) arrangement -> 128x128 block tile. No divergent returns: load rows
// are clamped, stores masked, so EXEC is all-ones at every WMMA.
__global__ __launch_bounds__(256)
void gcn_wmma_gemm_kernel(const ushort_t* __restrict__ A, const ushort_t* __restrict__ Bt,
                          float* __restrict__ C, const float* __restrict__ bias,
                          int M, int ldc) {
    const int K = HH;
    int lane  = threadIdx.x & 31;
    int wave  = threadIdx.x >> 5;
    int waveM = wave & 3;                       // 0..3
    int waveN = wave >> 2;                      // 0..1
    int r  = lane & 15;
    int hk = lane >> 4;
    int rowBase = blockIdx.x * 128 + waveM * 32;    // 2 m-tiles of 16
    int colBase = blockIdx.y * 128 + waveN * 64;    // 4 n-tiles of 16

    // A fragment lane: row = r (+16 for 2nd tile); dwords0-3: K=hk*8+0..7, dwords4-7: K=16+hk*8+0..7
    int ra0 = rowBase + r;       if (ra0 > M - 1) ra0 = M - 1;   // clamp (store is masked)
    int ra1 = rowBase + 16 + r;  if (ra1 > M - 1) ra1 = M - 1;
    const ushort_t* ap0 = A + (size_t)ra0 * K + hk * 8;
    const ushort_t* ap1 = A + (size_t)ra1 * K + hk * 8;
    // B fragment lane: col = r (+16n); dwords0-3: K=hk*8+0..7, dwords4-7: K=16+hk*8+0..7
    const ushort_t* bp0 = Bt + (size_t)(colBase + r)      * K + hk * 8;
    const ushort_t* bp1 = Bt + (size_t)(colBase + 16 + r) * K + hk * 8;
    const ushort_t* bp2 = Bt + (size_t)(colBase + 32 + r) * K + hk * 8;
    const ushort_t* bp3 = Bt + (size_t)(colBase + 48 + r) * K + hk * 8;

    v8f acc[2][4] = {};
    #pragma unroll 2
    for (int k0 = 0; k0 < K; k0 += 32) {
        FragU a0, a1, b0, b1, b2, b3;
        a0.q[0] = *(const int4*)(ap0 + k0);  a0.q[1] = *(const int4*)(ap0 + k0 + 16);
        a1.q[0] = *(const int4*)(ap1 + k0);  a1.q[1] = *(const int4*)(ap1 + k0 + 16);
        b0.q[0] = *(const int4*)(bp0 + k0);  b0.q[1] = *(const int4*)(bp0 + k0 + 16);
        b1.q[0] = *(const int4*)(bp1 + k0);  b1.q[1] = *(const int4*)(bp1 + k0 + 16);
        b2.q[0] = *(const int4*)(bp2 + k0);  b2.q[1] = *(const int4*)(bp2 + k0 + 16);
        b3.q[0] = *(const int4*)(bp3 + k0);  b3.q[1] = *(const int4*)(bp3 + k0 + 16);
        __builtin_prefetch(ap0 + k0 + 64, 0, 1);    // global_prefetch next K panel
        __builtin_prefetch(ap1 + k0 + 64, 0, 1);
        __builtin_prefetch(bp0 + k0 + 64, 0, 1);
        __builtin_prefetch(bp2 + k0 + 64, 0, 1);
        acc[0][0] = __builtin_amdgcn_wmma_f32_16x16x32_bf16(false, a0.v, false, b0.v, (short)0, acc[0][0], false, false);
        acc[0][1] = __builtin_amdgcn_wmma_f32_16x16x32_bf16(false, a0.v, false, b1.v, (short)0, acc[0][1], false, false);
        acc[0][2] = __builtin_amdgcn_wmma_f32_16x16x32_bf16(false, a0.v, false, b2.v, (short)0, acc[0][2], false, false);
        acc[0][3] = __builtin_amdgcn_wmma_f32_16x16x32_bf16(false, a0.v, false, b3.v, (short)0, acc[0][3], false, false);
        acc[1][0] = __builtin_amdgcn_wmma_f32_16x16x32_bf16(false, a1.v, false, b0.v, (short)0, acc[1][0], false, false);
        acc[1][1] = __builtin_amdgcn_wmma_f32_16x16x32_bf16(false, a1.v, false, b1.v, (short)0, acc[1][1], false, false);
        acc[1][2] = __builtin_amdgcn_wmma_f32_16x16x32_bf16(false, a1.v, false, b2.v, (short)0, acc[1][2], false, false);
        acc[1][3] = __builtin_amdgcn_wmma_f32_16x16x32_bf16(false, a1.v, false, b3.v, (short)0, acc[1][3], false, false);
    }

    // C layout per tile: VGPR i -> row = tileRow + hk*8 + i, col = tileCol + (lane&15)
    float bv[4]; int cv[4]; bool cok[4];
    #pragma unroll
    for (int n = 0; n < 4; ++n) {
        cv[n]  = colBase + n * 16 + r;
        cok[n] = cv[n] < ldc;
        bv[n]  = (bias && cok[n]) ? bias[cv[n]] : 0.f;
    }
    #pragma unroll
    for (int m = 0; m < 2; ++m) {
        int rb = rowBase + m * 16 + hk * 8;
        #pragma unroll
        for (int i = 0; i < 8; ++i) {
            int row = rb + i;
            if (row < M) {
                #pragma unroll
                for (int n = 0; n < 4; ++n)
                    if (cok[n]) C[(size_t)row * ldc + cv[n]] = acc[m][n][i] + bv[n];
            }
        }
    }
}

// ---------- per-column sum / sumsq over node dim (for BatchNorm stats) ----------
__global__ __launch_bounds__(256)
void gcn_colstats_kernel(const float* __restrict__ X, float* __restrict__ st, int N) {
    int c  = threadIdx.x;                      // columns c and c+256
    int r0 = blockIdx.x * 512;
    int r1 = r0 + 512; if (r1 > N) r1 = N;
    float s0 = 0.f, q0 = 0.f, s1 = 0.f, q1 = 0.f;
    for (int r = r0; r < r1; ++r) {
        float a = X[(size_t)r * HH + c];        s0 += a; q0 += a * a;
        float b = X[(size_t)r * HH + c + 256];  s1 += b; q1 += b * b;
    }
    atomicAdd(&st[c],            s0); atomicAdd(&st[HH + c],       q0);
    atomicAdd(&st[c + 256],      s1); atomicAdd(&st[HH + c + 256], q1);
}

// ---------- fused layer-1 epilogue: relu( BN(outer(s,W1)) + BN(U1) ) -> bf16 ----------
__global__ __launch_bounds__(256)
void gcn_ew1_kernel(const float* __restrict__ s, const float* __restrict__ scal2,
                    const float* __restrict__ W1, const float* __restrict__ g1a,
                    const float* __restrict__ be1a, const float* __restrict__ U1,
                    const float* __restrict__ cs1, const float* __restrict__ g1b,
                    const float* __restrict__ be1b, ushort_t* __restrict__ hbf, int N) {
    int r = blockIdx.x;
    float invN = 1.f / (float)N;
    float ms = scal2[0] * invN;
    float vs = scal2[1] * invN - ms * ms;
    float sv = s[r] - ms;
    for (int c = threadIdx.x; c < HH; c += 256) {
        float w  = W1[c];
        float a  = (sv * w) * rsqrtf(vs * w * w + EPSB) * g1a[c] + be1a[c]; // closed-form BN of rank-1 GCN
        float m  = cs1[c] * invN;
        float vv = cs1[HH + c] * invN - m * m;
        float u  = (U1[(size_t)r * HH + c] - m) * rsqrtf(vv + EPSB) * g1b[c] + be1b[c];
        float h  = a + u;
        h = h > 0.f ? h : 0.f;
        hbf[(size_t)r * HH + c] = f32_bf16(h);
    }
}

// ---------- layer-2 edge scatter: agg[dst,:] += V[src,:] ----------
__global__ __launch_bounds__(256)
void gcn_scatter_rows_kernel(const float* __restrict__ V, const int* __restrict__ src,
                             const int* __restrict__ dst, float* __restrict__ agg) {
    int e = blockIdx.x;
    const float* vp  = V   + (size_t)src[e] * HH;
    float*       apg = agg + (size_t)dst[e] * HH;
    for (int c = threadIdx.x; c < HH; c += 256) atomicAdd(&apg[c], vp[c]);
}

// ---------- fused layer-2 epilogue: relu( BN(agg2) + BN(U2) ) -> bf16 ----------
__global__ __launch_bounds__(256)
void gcn_ew2_kernel(const float* __restrict__ agg, const float* __restrict__ cs2,
                    const float* __restrict__ g2a, const float* __restrict__ be2a,
                    const float* __restrict__ U2, const float* __restrict__ cs3,
                    const float* __restrict__ g2b, const float* __restrict__ be2b,
                    ushort_t* __restrict__ hbf, int N) {
    int r = blockIdx.x;
    float invN = 1.f / (float)N;
    for (int c = threadIdx.x; c < HH; c += 256) {
        float ma = cs2[c] * invN;
        float va = cs2[HH + c] * invN - ma * ma;
        float a  = (agg[(size_t)r * HH + c] - ma) * rsqrtf(va + EPSB) * g2a[c] + be2a[c];
        float mb = cs3[c] * invN;
        float vb = cs3[HH + c] * invN - mb * mb;
        float u  = (U2[(size_t)r * HH + c] - mb) * rsqrtf(vb + EPSB) * g2b[c] + be2b[c];
        float h  = a + u;
        h = h > 0.f ? h : 0.f;
        hbf[(size_t)r * HH + c] = f32_bf16(h);
    }
}

// ==================== host side ====================
extern "C" void kernel_launch(void* const* d_in, const int* in_sizes, int n_in,
                              void* d_out, int out_size, void* d_ws, size_t ws_size,
                              hipStream_t stream) {
    (void)n_in; (void)out_size;
    const float* x    = (const float*)d_in[0];
    const int*   ei   = (const int*)  d_in[1];
    const float* alt  = (const float*)d_in[2];
    const float* W1   = (const float*)d_in[3];
    const float* g1a  = (const float*)d_in[5];
    const float* be1a = (const float*)d_in[6];
    const float* A1a  = (const float*)d_in[7];
    const float* A1b  = (const float*)d_in[9];
    const float* g1b  = (const float*)d_in[11];
    const float* be1b = (const float*)d_in[12];
    const float* W2   = (const float*)d_in[13];
    const float* g2a  = (const float*)d_in[15];
    const float* be2a = (const float*)d_in[16];
    const float* A2a  = (const float*)d_in[17];
    const float* A2b  = (const float*)d_in[19];
    const float* g2b  = (const float*)d_in[21];
    const float* be2b = (const float*)d_in[22];
    const float* Wl   = (const float*)d_in[23];
    const float* bl   = (const float*)d_in[24];

    int N = in_sizes[0];                 // 50000
    int E = in_sizes[1] / 2;             // 500000
    const int* src = ei;
    const int* dst = ei + E;

    // ---- carve workspace ----
    char* p = (char*)d_ws;
    auto alloc = [&](size_t bytes) -> char* {
        char* q = p; p += (bytes + 255) & ~(size_t)255; return q;
    };
    float*    s     = (float*)   alloc((size_t)N * 4);
    float*    scal2 = (float*)   alloc(256);
    float*    cs1   = (float*)   alloc(2 * HH * 4);
    float*    cs2   = (float*)   alloc(2 * HH * 4);
    float*    cs3   = (float*)   alloc(2 * HH * 4);
    ushort_t* BT1   = (ushort_t*)alloc((size_t)HH * HH * 2);       // A1b^T bf16
    ushort_t* BT2   = (ushort_t*)alloc((size_t)HH * HH * 2);       // W2^T  bf16
    ushort_t* BT3   = (ushort_t*)alloc((size_t)HH * HH * 2);       // A2b^T bf16
    ushort_t* BT4   = (ushort_t*)alloc((size_t)OUTPAD * HH * 2);   // Wl^T  bf16, zero-padded
    float*    bufA  = (float*)   alloc((size_t)N * HH * 4);        // U1, then agg2
    float*    bufB  = (float*)   alloc((size_t)N * HH * 4);        // V2, then U2
    ushort_t* tbf   = (ushort_t*)alloc((size_t)N * HH * 2);        // T1, then T2
    ushort_t* hbf   = (ushort_t*)alloc((size_t)N * HH * 2);        // h1, then h2
    if ((size_t)(p - (char*)d_ws) > ws_size) return;               // deterministic guard

    // ---- zero accumulators (capture-safe memset nodes) ----
    hipMemsetAsync(s,     0, (size_t)N * 4, stream);
    hipMemsetAsync(scal2, 0, 256,           stream);
    hipMemsetAsync(cs1,   0, 2 * HH * 4,    stream);
    hipMemsetAsync(cs2,   0, 2 * HH * 4,    stream);
    hipMemsetAsync(cs3,   0, 2 * HH * 4,    stream);

    // ---- weights -> transposed bf16 ----
    gcn_convT_kernel<<<HH,     256, 0, stream>>>(A1b, BT1, HH);
    gcn_convT_kernel<<<HH,     256, 0, stream>>>(W2,  BT2, HH);
    gcn_convT_kernel<<<HH,     256, 0, stream>>>(A2b, BT3, HH);
    gcn_convT_kernel<<<OUTPAD, 256, 0, stream>>>(Wl,  BT4, OUTC);

    dim3 gemm_grid((unsigned)((N + 127) / 128), HH / 128);     // 128x128 block tiles
    dim3 gemm_grid_out((unsigned)((N + 127) / 128), OUTPAD / 128);

    // ---- layer 1 ----
    gcn_scatter_scalar_kernel<<<(E + 255) / 256, 256, 0, stream>>>(x, src, dst, s, E);
    gcn_reduce_scalar_kernel<<<64, 256, 0, stream>>>(s, scal2, N);
    gcn_alter_mm_kernel<<<N, 256, 0, stream>>>(alt, A1a, tbf);                         // T1
    gcn_wmma_gemm_kernel<<<gemm_grid, 256, 0, stream>>>(tbf, BT1, bufA,
                                                        nullptr, N, HH);               // U1
    gcn_colstats_kernel<<<(N + 511) / 512, 256, 0, stream>>>(bufA, cs1, N);
    gcn_ew1_kernel<<<N, 256, 0, stream>>>(s, scal2, W1, g1a, be1a,
                                          bufA, cs1, g1b, be1b, hbf, N);               // h1 (bf16)

    // ---- layer 2 ----
    gcn_wmma_gemm_kernel<<<gemm_grid, 256, 0, stream>>>(hbf, BT2, bufB,
                                                        nullptr, N, HH);               // V2 = h1@W2
    hipMemsetAsync(bufA, 0, (size_t)N * HH * 4, stream);                               // agg2 = 0
    gcn_scatter_rows_kernel<<<E, 256, 0, stream>>>(bufB, src, dst, bufA);              // agg2
    gcn_colstats_kernel<<<(N + 511) / 512, 256, 0, stream>>>(bufA, cs2, N);
    gcn_alter_mm_kernel<<<N, 256, 0, stream>>>(alt, A2a, tbf);                         // T2
    gcn_wmma_gemm_kernel<<<gemm_grid, 256, 0, stream>>>(tbf, BT3, bufB,
                                                        nullptr, N, HH);               // U2
    gcn_colstats_kernel<<<(N + 511) / 512, 256, 0, stream>>>(bufB, cs3, N);
    gcn_ew2_kernel<<<N, 256, 0, stream>>>(bufA, cs2, g2a, be2a,
                                          bufB, cs3, g2b, be2b, hbf, N);               // h2 (bf16)

    // ---- output head: d_out = h2 @ Wl + bl ----
    gcn_wmma_gemm_kernel<<<gemm_grid_out, 256, 0, stream>>>(hbf, BT4,
                                                            (float*)d_out, bl,
                                                            N, OUTC);
}
